// DilatedConv1D_18313740550563
// MI455X (gfx1250) — matrix operand order repaired and checked by reading, not compiled, not run on hardware
//
#include <hip/hip_runtime.h>

typedef __attribute__((ext_vector_type(16))) _Float16     v16h;
typedef __attribute__((ext_vector_type(8)))  float        v8f;
typedef __attribute__((ext_vector_type(8)))  unsigned int v8u;
typedef __attribute__((ext_vector_type(4)))  unsigned int v4u;
typedef __attribute__((ext_vector_type(2)))  float        v2f;

#define NC   128
#define NT   32768
#define NOUT 128
#define NP   (NT + 1)      /* 32769 output positions */
#define NCOLS 18           /* staged x columns per 16-wide p-tile (16 + 2 dil) */
#define LPITCH 68          /* LDS row pitch in dwords: 64 cpairs + 4 pad      */
#define TPB  17            /* p-tiles per block */
#define NPG  128           /* p-groups: 128*17 = 2176 >= 2049 tiles */

__device__ __forceinline__ unsigned int pk(float a, float b) {
    auto h = __builtin_amdgcn_cvt_pkrtz(a, b);   /* v2 __fp16, a -> [15:0] */
    return __builtin_bit_cast(unsigned int, h);
}

__global__ __launch_bounds__(256) void dconv_wmma(
        const float* __restrict__ x, const float* __restrict__ W,
        const float* __restrict__ bias, float* __restrict__ out)
{
    /* staged x tile: packed f16 (c, c+1) pairs, transposed: lds[q][cpair] */
    __shared__ __align__(16) unsigned int lds[NCOLS * LPITCH];

    const int tid  = threadIdx.x;
    const int og   = tid >> 5;         /* wave index == o-tile 0..7 */
    const int lane = tid & 31;
    const int hf   = lane >> 4;
    const int lm   = lane & 15;

    const int b = blockIdx.x & 7;      /* batch */
    const int g = blockIdx.x >> 3;     /* p-group 0..127 */

    const float* xb   = x   + (size_t)b * NC * NT;
    float*       outb = out + (size_t)b * NOUT * NP;

    const int o_lane = og * 16 + lm;   /* A-matrix row for this lane */

    /* ---- W fragments: a0/a1 = tap0/tap1 per 32-wide c-chunk.
       A 16x32 f16: lanes 0-15 K={0..15}, lanes 16-31 K={16..31};
       element e: K = hf*16 + e. ---- */
    v16h a0[4], a1[4];
    #pragma unroll
    for (int cc = 0; cc < 4; ++cc) {
        v8u f0, f1;
        #pragma unroll
        for (int ep = 0; ep < 8; ++ep) {          /* e = 2*ep -> c, c+1 */
            const int e    = 2 * ep;
            const int kidx = hf * 16 + e;
            const int c    = cc * 32 + kidx;
            v2f wA = *(const v2f*)(W + ((size_t)o_lane * NC + c) * 2);
            v2f wB = *(const v2f*)(W + ((size_t)o_lane * NC + c + 1) * 2);
            f0[ep] = pk(wA.x, wB.x);              /* tap 0 */
            f1[ep] = pk(wA.y, wB.y);              /* tap 1 */
        }
        a0[cc] = __builtin_bit_cast(v16h, f0);
        a1[cc] = __builtin_bit_cast(v16h, f1);
    }

    /* ---- bias per accumulator row (C row M = r + hf*8) ---- */
    float bv[8];
    #pragma unroll
    for (int r = 0; r < 8; ++r)
        bv[r] = bias[og * 16 + hf * 8 + r];

    /* ---- p-tile loop (uniform across block: barriers are safe) ---- */
    for (int t = 0; t < TPB; ++t) {
        const int p0 = (g * TPB + t) * 16;
        if (p0 >= NP) break;

        __syncthreads();               /* previous tile's LDS reads done */

        /* ---- stage x[0..127][p0..p0+17] as packed f16 pairs, transposed.
           item -> cpair = item/18, q = item%18 (q fastest => coalesced). */
        #pragma unroll
        for (int j = 0; j < 5; ++j) {
            const int item = tid + j * 256;
            if (item < 64 * NCOLS) {
                const int cp = item / NCOLS;
                const int q  = item - cp * NCOLS;
                const int qg = p0 + q;
                const unsigned qa = (qg < NT) ? (unsigned)qg : (unsigned)(NT - 1);
                float lo = xb[(unsigned)(2 * cp) * NT + qa];
                float hi = xb[(unsigned)(2 * cp + 1) * NT + qa];
                if (qg >= NT) { lo = 0.0f; hi = 0.0f; }   /* zero pad */
                lds[q * LPITCH + cp] = pk(lo, hi);
            }
        }

        __syncthreads();

        /* ---- compute: B fragment = 8 consecutive LDS dwords at
           row q = lm + 2k, cpair base = cc*16 + hf*8 (2x ds_load_b128) ---- */
        v8f acc = {};
        #pragma unroll
        for (int cc = 0; cc < 4; ++cc) {
            #pragma unroll
            for (int k = 0; k < 2; ++k) {
                const int a  = (lm + 2 * k) * LPITCH + cc * 16 + hf * 8;
                const v4u lo4 = *(const v4u*)&lds[a];
                const v4u hi4 = *(const v4u*)&lds[a + 4];
                v8u f;
                f[0] = lo4[0]; f[1] = lo4[1]; f[2] = lo4[2]; f[3] = lo4[3];
                f[4] = hi4[0]; f[5] = hi4[1]; f[6] = hi4[2]; f[7] = hi4[3];
                const v16h bf = __builtin_bit_cast(v16h, f);
                acc = __builtin_amdgcn_wmma_f32_16x16x32_f16(
                        false, (k == 0) ? a0[cc] : a1[cc], false, bf,
                        (short)0, acc, false, false);
            }
        }

        /* ---- bias + store: C VGPR r -> row r + hf*8, col lm ---- */
        const int p = p0 + lm;
        if (p < NP) {
            #pragma unroll
            for (int r = 0; r < 8; ++r) {
                const int o = og * 16 + hf * 8 + r;
                outb[(unsigned)o * NP + (unsigned)p] = acc[r] + bv[r];
            }
        }
    }
}

extern "C" void kernel_launch(void* const* d_in, const int* in_sizes, int n_in,
                              void* d_out, int out_size, void* d_ws, size_t ws_size,
                              hipStream_t stream) {
    const float* x = (const float*)d_in[0];
    const float* W = (const float*)d_in[1];
    const float* b = (const float*)d_in[2];
    float* out = (float*)d_out;
    dim3 grid(8 * NPG);     /* 8 batches x 128 p-groups = 1024 blocks */
    dim3 block(256);        /* 8 waves: one o-tile each */
    hipLaunchKernelGGL(dconv_wmma, grid, block, 0, stream, x, W, b, out);
}
